// SpectralConv1d_86397562126834
// MI455X (gfx1250) — matrix operand order repaired and checked by reading, not compiled, not run on hardware
//
#include <hip/hip_runtime.h>

// Problem constants (from reference)
#define B_SZ   16
#define CIN    64
#define COUT   64
#define NPTS   4096
#define MODES  32
#define NF     2049              // NPTS/2 + 1
#define NFP    2064              // NF padded up to multiple of 16

typedef __attribute__((ext_vector_type(16))) __bf16 v16bf;
typedef __attribute__((ext_vector_type(8)))  __bf16 v8bf;
typedef __attribute__((ext_vector_type(8)))  float  v8f;

// ---- workspace layout (bytes); total ~9.6 MB ----
// WfT : bf16 [64][4096]   forward twiddles, row j<32: cos(2pi n j/N), j>=32: -sin
// W2T : bf16 [2064][64]   inverse twiddles, col k<32: cos(2pi k n/nf)/nf, k>=32: -sin/nf
// X   : f32  [1024][64]   spectrum (Re cols 0..31, Im cols 32..63)
// A2  : bf16 [1024][64]   mixed modes (Or cols 0..31, Oi cols 32..63)
// XB  : bf16 [1024][4096] bf16 copy of x (one-pass convert; removes VALU from GEMM loop)
#define OFF_WFT 0u
#define OFF_W2T (OFF_WFT + 64u * 4096u * 2u)          // 524288
#define OFF_X   (OFF_W2T + (unsigned)NFP * 64u * 2u)  // 788480
#define OFF_A2  (OFF_X   + 1024u * 64u * 4u)          // 1050624
#define OFF_XB  (OFF_A2  + 1024u * 64u * 2u)          // 1181696 (+8 MB)

#define TWO_PI 6.28318530717958647692f

static __device__ __forceinline__ __bf16 f2bf(float f) {
    unsigned u = __builtin_bit_cast(unsigned, f);
    unsigned r = u + 0x7FFFu + ((u >> 16) & 1u);     // round-to-nearest-even
    return __builtin_bit_cast(__bf16, (unsigned short)(r >> 16));
}

// ---------------- x -> bf16 one-pass convert ----------------
__global__ void __launch_bounds__(256) xcast(const float* __restrict__ x,
                                             __bf16* __restrict__ xb) {
    int t = blockIdx.x * 256 + threadIdx.x;          // 524288 threads, 8 elems each
    const float* p = x + (size_t)t * 8;
    float4 f0 = *(const float4*)(p);
    float4 f1 = *(const float4*)(p + 4);
    v8bf o;
    o[0] = f2bf(f0.x); o[1] = f2bf(f0.y); o[2] = f2bf(f0.z); o[3] = f2bf(f0.w);
    o[4] = f2bf(f1.x); o[5] = f2bf(f1.y); o[6] = f2bf(f1.z); o[7] = f2bf(f1.w);
    *(v8bf*)(xb + (size_t)t * 8) = o;
}

// ---------------- twiddle init ----------------
__global__ void __launch_bounds__(256) init_wft(__bf16* __restrict__ wft) {
    int idx = blockIdx.x * 256 + threadIdx.x;        // 64*4096 threads
    int j = idx >> 12;                               // 0..63 (spectral column)
    int n = idx & 4095;                              // 0..4095 (time index)
    int m = (j < MODES) ? j : (j - MODES);
    int t = (m * n) & (NPTS - 1);                    // phase mod N, exact in int
    float th = (TWO_PI / (float)NPTS) * (float)t;
    float v = (j < MODES) ? __cosf(th) : -__sinf(th);
    wft[idx] = f2bf(v);
}

__global__ void __launch_bounds__(256) init_w2t(__bf16* __restrict__ w2t) {
    int idx = blockIdx.x * 256 + threadIdx.x;        // 2064*64 threads
    int n = idx >> 6;                                // 0..2063 (output sample)
    int j = idx & 63;                                // 0..63 (mode column)
    float v = 0.0f;
    if (n < NF) {
        int m = (j < MODES) ? j : (j - MODES);
        int t = (m * n) % NF;                        // phase mod nf
        float th = (TWO_PI / (float)NF) * (float)t;
        v = ((j < MODES) ? __cosf(th) : -__sinf(th)) * (1.0f / (float)NF);
    }
    w2t[idx] = f2bf(v);
}

// ---------------- stage 1: truncated forward DFT (bf16 WMMA, f32 accum) ----------------
// X[1024 x 64] = xb[1024 x 4096] @ WfT^T ; 64 blocks (M-tiles) x 4 waves (N-tiles)
// Inner loop: 4x global_load_b128 + 1x v_wmma; two accumulators break the C->C chain.
__global__ void __launch_bounds__(128) dft_fwd_wmma(const __bf16* __restrict__ xb,
                                                    const __bf16* __restrict__ wft,
                                                    float* __restrict__ Xout) {
    const int lane  = threadIdx.x & 31;
    const int ntile = threadIdx.x >> 5;      // 0..3
    const int mtile = blockIdx.x;            // 0..63
    const int hi    = lane >> 4;             // half-wave select
    const int l15   = lane & 15;

    const __bf16* arow = xb  + (size_t)(mtile * 16 + l15) * NPTS;
    const __bf16* brow = wft + (size_t)(ntile * 16 + l15) * NPTS;

    v8f acc0 = {};
    v8f acc1 = {};
    for (int kk = 0; kk < NPTS; kk += 64) {
        // --- chunk 0 (K = kk .. kk+31) ---
        {
            int ab = kk + hi * 16;
            v8bf a0 = *(const v8bf*)(arow + ab);
            v8bf a1 = *(const v8bf*)(arow + ab + 8);
            v16bf a;
#pragma unroll
            for (int e = 0; e < 8; ++e) { a[e] = a0[e]; a[e + 8] = a1[e]; }
            v16bf b = *(const v16bf*)(brow + kk + hi * 16);
            acc0 = __builtin_amdgcn_wmma_f32_16x16x32_bf16(false, a, false, b,
                                                           (short)0, acc0, false, false);
        }
        // --- chunk 1 (K = kk+32 .. kk+63), independent accumulator ---
        {
            int ab = kk + 32 + hi * 16;
            v8bf a0 = *(const v8bf*)(arow + ab);
            v8bf a1 = *(const v8bf*)(arow + ab + 8);
            v16bf a;
#pragma unroll
            for (int e = 0; e < 8; ++e) { a[e] = a0[e]; a[e + 8] = a1[e]; }
            v16bf b = *(const v16bf*)(brow + kk + 32 + hi * 16);
            acc1 = __builtin_amdgcn_wmma_f32_16x16x32_bf16(false, a, false, b,
                                                           (short)0, acc1, false, false);
        }
    }
    // D layout: VGPR r -> row (hi*8 + r), lane&15 -> column
    const int col = ntile * 16 + l15;
    const int rowbase = mtile * 16 + hi * 8;
#pragma unroll
    for (int r = 0; r < 8; ++r)
        Xout[(rowbase + r) * 64 + col] = acc0[r] + acc1[r];
}

// ---------------- stage 2: per-mode complex channel mixing (tiny) ----------------
__global__ void __launch_bounds__(256) mode_mix(const float* __restrict__ X,
                                                const float* __restrict__ w1,
                                                __bf16* __restrict__ a2) {
    int t = blockIdx.x * 256 + threadIdx.x;  // 16*64*32 = 32768 threads
    int k = t & 31;
    int o = (t >> 5) & 63;
    int b = t >> 11;
    float orr = 0.0f, oii = 0.0f;
    for (int i = 0; i < CIN; ++i) {
        const float* xp = X + (size_t)(b * CIN + i) * 64;
        float xr = xp[k];
        float xi = xp[32 + k];
        const float* wp = w1 + ((size_t)(i * COUT + o) * MODES + k) * 2;
        float wr = wp[0], wi = wp[1];
        orr += xr * wr - xi * wi;
        oii += xr * wi + xi * wr;
    }
    int m = b * COUT + o;
    a2[m * 64 + k]      = f2bf(orr);
    a2[m * 64 + 32 + k] = f2bf(oii);
}

// ---------------- stage 3: inverse transform (bf16 WMMA, f32 accum) ----------------
// y[1024 x 2049] = A2[1024 x 64] @ W2T^T ; 64 x 129 tiles, one wave per tile
__global__ void __launch_bounds__(256) idft_wmma(const __bf16* __restrict__ a2,
                                                 const __bf16* __restrict__ w2t,
                                                 float* __restrict__ y) {
    const int wid   = blockIdx.x * 8 + (threadIdx.x >> 5); // exactly 8256 waves
    const int mtile = wid / 129;
    const int ntile = wid % 129;
    const int lane  = threadIdx.x & 31;
    const int hi    = lane >> 4;
    const int l15   = lane & 15;

    const __bf16* arow = a2  + (size_t)(mtile * 16 + l15) * 64;
    const __bf16* brow = w2t + (size_t)(ntile * 16 + l15) * 64;

    v8f acc = {};
#pragma unroll
    for (int kk = 0; kk < 64; kk += 32) {
        v8bf a0 = *(const v8bf*)(arow + kk + hi * 16);
        v8bf a1 = *(const v8bf*)(arow + kk + hi * 16 + 8);
        v16bf a;
#pragma unroll
        for (int e = 0; e < 8; ++e) { a[e] = a0[e]; a[e + 8] = a1[e]; }
        v16bf b = *(const v16bf*)(brow + kk + hi * 16);
        acc = __builtin_amdgcn_wmma_f32_16x16x32_bf16(false, a, false, b,
                                                      (short)0, acc, false, false);
    }
    const int n = ntile * 16 + l15;
    if (n < NF) {                        // tail mask only at the store; WMMA ran with full EXEC
        const int rowbase = mtile * 16 + hi * 8;
#pragma unroll
        for (int r = 0; r < 8; ++r)
            y[(size_t)(rowbase + r) * NF + n] = acc[r];
    }
}

extern "C" void kernel_launch(void* const* d_in, const int* in_sizes, int n_in,
                              void* d_out, int out_size, void* d_ws, size_t ws_size,
                              hipStream_t stream) {
    const float* x  = (const float*)d_in[0];   // [16,64,4096] f32
    const float* w1 = (const float*)d_in[1];   // [64,64,32,2] f32
    float* y        = (float*)d_out;           // [16,64,2049] f32

    char* ws = (char*)d_ws;
    __bf16* wft = (__bf16*)(ws + OFF_WFT);
    __bf16* w2t = (__bf16*)(ws + OFF_W2T);
    float*  X   = (float*) (ws + OFF_X);
    __bf16* a2  = (__bf16*)(ws + OFF_A2);
    __bf16* xb  = (__bf16*)(ws + OFF_XB);

    // twiddle tables + bf16 copy of x (deterministic, regenerated every call)
    init_wft<<<(64 * 4096) / 256, 256, 0, stream>>>(wft);
    init_w2t<<<(NFP * 64) / 256, 256, 0, stream>>>(w2t);
    xcast<<<(1024 * NPTS / 8) / 256, 256, 0, stream>>>(x, xb);

    // stage 1: forward DFT -> X  (64 M-tiles, 4 N-tile waves each)
    dft_fwd_wmma<<<64, 128, 0, stream>>>(xb, wft, X);

    // stage 2: complex mode mixing -> A2
    mode_mix<<<(B_SZ * COUT * MODES) / 256, 256, 0, stream>>>(X, w1, a2);

    // stage 3: inverse transform -> y  (64*129 tiles / 8 waves per block)
    idft_wmma<<<(64 * 129) / 8, 256, 0, stream>>>(a2, w2t, y);
}
